// AttentionPoolingTimesteps_56770877718768
// MI455X (gfx1250) — compile-verified, matrix operand order checked by reading, not run on hardware
//
#include <hip/hip_runtime.h>
#include <hip/hip_bf16.h>

// ---- CDNA5 WMMA types ------------------------------------------------------
typedef __attribute__((ext_vector_type(16))) __bf16 v16bf;
typedef __attribute__((ext_vector_type(8)))  float  v8f;

#define T_DIM 128
#define C_DIM 256
#define XSTR  264   // ushort stride for q/k tiles: 528B rows -> 4-bank rotate, conflict-free b128
#define WSTR  40    // ushort stride for W strip: 80B rows, 16B aligned, 20-dword rotate (conflict-free)

static __device__ __forceinline__ unsigned short f2bf(float f) {
    unsigned int u = __float_as_uint(f);
    u += 0x7FFFu + ((u >> 16) & 1u);          // round-to-nearest-even
    return (unsigned short)(u >> 16);
}
static __device__ __forceinline__ float bf2f(unsigned short h) {
    return __uint_as_float(((unsigned int)h) << 16);
}

// Load one 16x32 bf16 fragment from LDS (A layout; B is loaded as B^T in the
// same layout). off must be a multiple of 8 ushorts (16B).
static __device__ __forceinline__ v16bf load_frag(const unsigned short* p, int off) {
    union { v16bf v; uint4 q[2]; } u;
    u.q[0] = *(const uint4*)(p + off);
    u.q[1] = *(const uint4*)(p + off + 16);
    return u.v;
}

// Build hi/lo bf16 fragments straight from global fp32 (this lane's 16 elements:
// two runs of 8 consecutive floats, second run at +16 columns).
static __device__ __forceinline__ void frag_from_global(const float* __restrict__ p,
                                                        v16bf* fh, v16bf* fl) {
    const float4* q4 = (const float4*)p;
    float4 A = q4[0], B = q4[1], C = q4[4], D = q4[5];
    float v[16] = {A.x, A.y, A.z, A.w, B.x, B.y, B.z, B.w,
                   C.x, C.y, C.z, C.w, D.x, D.y, D.z, D.w};
    union { v16bf bf; unsigned int u[8]; } H, L;
#pragma unroll
    for (int i = 0; i < 8; ++i) {
        unsigned short h0 = f2bf(v[2 * i]);
        unsigned short h1 = f2bf(v[2 * i + 1]);
        unsigned short l0 = f2bf(v[2 * i]     - bf2f(h0));
        unsigned short l1 = f2bf(v[2 * i + 1] - bf2f(h1));
        H.u[i] = (unsigned int)h0 | ((unsigned int)h1 << 16);
        L.u[i] = (unsigned int)l0 | ((unsigned int)l1 << 16);
    }
    *fh = H.bf;
    *fl = L.bf;
}

static __device__ __forceinline__ v8f wmma_bf16(v16bf a, v16bf b, v8f c) {
    // (neg_a, A, neg_b, B, c_mod, C, reuse_a, reuse_b)
    return __builtin_amdgcn_wmma_f32_16x16x32_bf16(false, a, false, b, (short)0, c, false, false);
}

// One [128x256] @ [256x256]^T projection in bf16x3 (hi*hi + lo*hi + hi*lo),
// f32 accumulate. Each wave owns a 16-row strip; output columns are processed
// in two passes of 8 j-tiles (64 accumulator VGPRs live -> stays <256 VGPRs).
// A-fragments come straight from global x (L2-resident); W strips are staged
// cooperatively per K-step. B fragments are ping-pong double-buffered so the
// DS->WMMA dependency becomes a partial s_wait_dscnt instead of a full drain.
static __device__ void do_proj(const float* __restrict__ xg_tile,
                               unsigned short* stg_hi, unsigned short* stg_lo,
                               const float* __restrict__ W, const float* __restrict__ bias,
                               unsigned short* dst_hi, unsigned short* dst_lo) {
    const int tid  = threadIdx.x;
    const int wv   = tid >> 5;        // wave 0..7 -> row strip base 16*wv
    const int lane = tid & 31;
    const int g    = lane >> 4;       // K-half select per ISA A/B layout
    const int ln   = lane & 15;
    const int row0 = wv * 16;

    for (int ph = 0; ph < 2; ++ph) {
        v8f acc[8];
        const v8f vzero = {0.f, 0.f, 0.f, 0.f, 0.f, 0.f, 0.f, 0.f};
#pragma unroll
        for (int j = 0; j < 8; ++j) acc[j] = vzero;

        for (int ks = 0; ks < 8; ++ks) {
            __syncthreads();
            // ---- cooperative stage of W[:, ks*32 .. ks*32+32) as bf16 hi/lo ----
            {
                const float4* wsrc = (const float4*)(W + tid * C_DIM + ks * 32);
                __builtin_prefetch(W + tid * C_DIM + ((ks + 1) & 7) * 32, 0, 1);
                const int wb = tid * WSTR;
#pragma unroll
                for (int i = 0; i < 8; ++i) {
                    float4 f = wsrc[i];
                    unsigned short h0 = f2bf(f.x), h1 = f2bf(f.y), h2 = f2bf(f.z), h3 = f2bf(f.w);
                    stg_hi[wb + i*4 + 0] = h0; stg_lo[wb + i*4 + 0] = f2bf(f.x - bf2f(h0));
                    stg_hi[wb + i*4 + 1] = h1; stg_lo[wb + i*4 + 1] = f2bf(f.y - bf2f(h1));
                    stg_hi[wb + i*4 + 2] = h2; stg_lo[wb + i*4 + 2] = f2bf(f.z - bf2f(h2));
                    stg_hi[wb + i*4 + 3] = h3; stg_lo[wb + i*4 + 3] = f2bf(f.w - bf2f(h3));
                }
            }
            __syncthreads();

            // A fragment for this (strip, ks) from global x
            v16bf a_hi, a_lo;
            frag_from_global(xg_tile + (row0 + ln) * C_DIM + ks * 32 + g * 8, &a_hi, &a_lo);

            // ping-pong pipelined B fragments
            const int bbase0 = ((ph * 8) * 16 + ln) * WSTR + g * 8;
            v16bf bh = load_frag(stg_hi, bbase0);
            v16bf bl = load_frag(stg_lo, bbase0);
#pragma unroll
            for (int j = 0; j < 8; ++j) {
                const int jn     = (j < 7) ? j + 1 : j;          // clamp: stay in bounds
                const int bbasen = ((ph * 8 + jn) * 16 + ln) * WSTR + g * 8;
                v16bf bh_n = load_frag(stg_hi, bbasen);
                v16bf bl_n = load_frag(stg_lo, bbasen);
                acc[j] = wmma_bf16(a_hi, bh, acc[j]);
                acc[j] = wmma_bf16(a_lo, bh, acc[j]);
                acc[j] = wmma_bf16(a_hi, bl, acc[j]);
                bh = bh_n;
                bl = bl_n;
            }
        }

        // ---- add bias, split to hi/lo, store this pass's half-strip ----
        // (C/D layout: M = r + 8g, N = ln; own strip only -> race-free)
#pragma unroll
        for (int j = 0; j < 8; ++j) {
            const int jt = ph * 8 + j;
            const float b = bias[jt * 16 + ln];
#pragma unroll
            for (int r = 0; r < 8; ++r) {
                float v = acc[j][r] + b;
                const int idx = (row0 + r + 8 * g) * XSTR + jt * 16 + ln;
                unsigned short h = f2bf(v);
                dst_hi[idx] = h;
                dst_lo[idx] = f2bf(v - bf2f(h));
            }
        }
    }
}

__global__ __launch_bounds__(256, 1)
void attnpool_kernel(const float* __restrict__ x, const unsigned char* __restrict__ mask,
                     const float* __restrict__ Wq, const float* __restrict__ bq,
                     const float* __restrict__ Wk, const float* __restrict__ bk,
                     const float* __restrict__ Wv, const float* __restrict__ bv,
                     float* __restrict__ out) {
    __shared__ __align__(16) unsigned short s_q_hi[T_DIM * XSTR];
    __shared__ __align__(16) unsigned short s_q_lo[T_DIM * XSTR];
    __shared__ __align__(16) unsigned short s_k_hi[T_DIM * XSTR];
    __shared__ __align__(16) unsigned short s_k_lo[T_DIM * XSTR];
    __shared__ __align__(16) unsigned short s_w_hi[C_DIM * WSTR];  // weight strip stage
    __shared__ __align__(16) unsigned short s_w_lo[C_DIM * WSTR];
    __shared__ __align__(16) float s_wd[T_DIM];                    // diagonal softmax weights
    __shared__ __align__(16) float s_y[C_DIM];                     // y = w^T x
    __shared__ float s_misc[2];                                    // count, sum_w

    const int bn   = blockIdx.x;            // 0 .. B*N-1
    const int tid  = threadIdx.x;
    const int wv   = tid >> 5;
    const int lane = tid & 31;
    const int g    = lane >> 4;
    const int ln   = lane & 15;
    const int row0 = wv * 16;

    const float* xg_tile = x + (size_t)bn * T_DIM * C_DIM;

    // ---- k = x@Wk^T + bk ; q = x@Wq^T + bq (A-frags streamed from global x) ----
    do_proj(xg_tile, s_w_hi, s_w_lo, Wk, bk, s_k_hi, s_k_lo);
    do_proj(xg_tile, s_w_hi, s_w_lo, Wq, bq, s_q_hi, s_q_lo);
    __syncthreads();

    // ---- scores strip S[16 x 128] = q_strip @ k^T (bf16x3), softmax diagonal ----
    {
        v8f sacc[8];
        const v8f vzero = {0.f, 0.f, 0.f, 0.f, 0.f, 0.f, 0.f, 0.f};
#pragma unroll
        for (int ct = 0; ct < 8; ++ct) sacc[ct] = vzero;

        for (int ks = 0; ks < 8; ++ks) {
            const int abase = (row0 + ln) * XSTR + ks * 32 + g * 8;
            v16bf a_hi = load_frag(s_q_hi, abase);
            v16bf a_lo = load_frag(s_q_lo, abase);

            const int kcol = ks * 32 + g * 8;
            v16bf bh = load_frag(s_k_hi, ln * XSTR + kcol);
            v16bf bl = load_frag(s_k_lo, ln * XSTR + kcol);
#pragma unroll
            for (int ct = 0; ct < 8; ++ct) {
                const int ctn    = (ct < 7) ? ct + 1 : ct;       // clamp: stay in bounds
                const int bbasen = (ctn * 16 + ln) * XSTR + kcol;
                v16bf bh_n = load_frag(s_k_hi, bbasen);
                v16bf bl_n = load_frag(s_k_lo, bbasen);
                sacc[ct] = wmma_bf16(a_hi, bh, sacc[ct]);
                sacc[ct] = wmma_bf16(a_lo, bh, sacc[ct]);
                sacc[ct] = wmma_bf16(a_hi, bl, sacc[ct]);
                bh = bh_n;
                bl = bl_n;
            }
        }

        const float SC = 0.0625f;   // 1/sqrt(256)
#pragma unroll
        for (int r = 0; r < 8; ++r) {
            float sv[8];
#pragma unroll
            for (int ct = 0; ct < 8; ++ct) sv[ct] = sacc[ct][r] * SC;
            float mx = sv[0];
#pragma unroll
            for (int ct = 1; ct < 8; ++ct) mx = fmaxf(mx, sv[ct]);
            // row max / sum over the 16-lane half (wave32: XOR 1,2,4,8 stays in-half)
            mx = fmaxf(mx, __shfl_xor(mx, 1, 32));
            mx = fmaxf(mx, __shfl_xor(mx, 2, 32));
            mx = fmaxf(mx, __shfl_xor(mx, 4, 32));
            mx = fmaxf(mx, __shfl_xor(mx, 8, 32));
            float sum = 0.f;
#pragma unroll
            for (int ct = 0; ct < 8; ++ct) sum += __expf(sv[ct] - mx);
            sum += __shfl_xor(sum, 1, 32);
            sum += __shfl_xor(sum, 2, 32);
            sum += __shfl_xor(sum, 4, 32);
            sum += __shfl_xor(sum, 8, 32);
            // diagonal element of row t = row0 + r + 8g lives in col-tile ct == wv,
            // in the lane with ln == r + 8g
            float d = __expf(sv[wv] - mx);
            if (ln == r + 8 * g) {
                const int t = row0 + r + 8 * g;
                const bool valid = mask[(size_t)bn * T_DIM + t] != 0;
                s_wd[t] = valid ? (d / sum) : 0.f;
            }
        }
    }
    __syncthreads();

    if (tid == 0) {
        float cnt = 0.f, sw = 0.f;
        for (int t = 0; t < T_DIM; ++t) {
            cnt += (mask[(size_t)bn * T_DIM + t] != 0) ? 1.f : 0.f;
            sw  += s_wd[t];
        }
        s_misc[0] = cnt;
        s_misc[1] = sw;
    }
    __syncthreads();

    // ---- y[c] = sum_t w[t] * x[t,c]  (x is L2-resident by now) ----
    {
        const float* xcol = xg_tile + tid;
        float acc = 0.f;
        for (int t = 0; t < T_DIM; ++t) acc += s_wd[t] * xcol[(size_t)t * C_DIM];
        s_y[tid] = acc;
    }
    __syncthreads();

    // ---- pooled[c] = (y . Wv[c,:] + sum_w * bv[c]) / (count + 1e-9) ----
    {
        const float cnt = s_misc[0];
        const float sw  = s_misc[1];
        const float4* wrow = (const float4*)(Wv + (size_t)tid * C_DIM);
        const float4* y4   = (const float4*)s_y;
        float dot = 0.f;
#pragma unroll 8
        for (int i = 0; i < 64; ++i) {
            float4 a = wrow[i], b = y4[i];
            dot += a.x * b.x + a.y * b.y + a.z * b.z + a.w * b.w;
        }
        float res = (cnt > 0.f) ? (dot + sw * bv[tid]) * (1.f / (cnt + 1e-9f)) : 0.f;
        out[(size_t)bn * C_DIM + tid] = res;
    }
}

extern "C" void kernel_launch(void* const* d_in, const int* in_sizes, int n_in,
                              void* d_out, int out_size, void* d_ws, size_t ws_size,
                              hipStream_t stream) {
    (void)in_sizes; (void)n_in; (void)out_size; (void)d_ws; (void)ws_size;
    const float*         x    = (const float*)d_in[0];
    const unsigned char* mask = (const unsigned char*)d_in[1];   // jax bool -> 1 byte
    const float*         Wq   = (const float*)d_in[2];
    const float*         bq   = (const float*)d_in[3];
    const float*         Wk   = (const float*)d_in[4];
    const float*         bk   = (const float*)d_in[5];
    const float*         Wv   = (const float*)d_in[6];
    const float*         bv   = (const float*)d_in[7];
    float*               out  = (float*)d_out;

    // one workgroup per (b,n) tile: 8*128 = 1024 blocks, 256 threads (8 waves, wave32)
    attnpool_kernel<<<dim3(1024), dim3(256), 0, stream>>>(x, mask, Wq, bq, Wk, bk, Wv, bv, out);
}
